// MultiModalMoE_420906795434
// MI455X (gfx1250) — compile-verified, matrix operand order checked by reading, not run on hardware
//
#include <hip/hip_runtime.h>
#include <hip/hip_bf16.h>

typedef __attribute__((ext_vector_type(16))) _Float16 v16h;
typedef __attribute__((ext_vector_type(8)))  _Float16 v8h;
typedef __attribute__((ext_vector_type(8)))  float    v8f;
typedef __attribute__((ext_vector_type(4)))  unsigned int u32x4;
typedef __attribute__((ext_vector_type(8)))  int      i32x8;
typedef __attribute__((ext_vector_type(4)))  int      i32x4;

#if __has_builtin(__builtin_amdgcn_tensor_load_to_lds) && \
    __has_builtin(__builtin_amdgcn_s_wait_tensorcnt)
#define USE_TDM 1
#else
#define USE_TDM 0
#endif

// Fragment element j (16-bit A/B, wave32):  K = 16*(j>>3) + 8*(lane>>4) + (j&7).
// => for fixed lane, j=0..7 contiguous at kb, j=8..15 contiguous at kb+16.
static __device__ __forceinline__ v16h ldfrag(const _Float16* __restrict__ p) {
  v8h lo = *(const v8h*)p;
  v8h hi = *(const v8h*)(p + 16);
  v16h r;
#pragma unroll
  for (int j = 0; j < 8; ++j) { r[j] = lo[j]; r[j + 8] = hi[j]; }
  return r;
}

static __device__ __forceinline__ v8f wmma16(v16h a, v16h b, v8f c) {
  return __builtin_amdgcn_wmma_f32_16x16x32_f16(false, a, false, b,
                                                (short)0, c, false, false);
}

#if USE_TDM
// Issue a TDM load of a 16-row x 32-elem f16 tile (row stride Kd elems) into LDS.
// D# per CDNA5 ISA ch.8: group0 = {flags, lds_addr, global_addr, type=2},
// group1 = {data_size=2B, tensor dims, tile dims, dim0 stride}. 2D -> groups 2/3/4 zero.
static __device__ __forceinline__ void tdm_load_tile(const _Float16* gsrc,
                                                     unsigned ldsAddr, int Kd) {
  unsigned long long ga = (unsigned long long)(size_t)gsrc;
  u32x4 g0;
  g0[0] = 1u;                                         // count=1, user mode
  g0[1] = ldsAddr;                                    // LDS byte address
  g0[2] = (unsigned)(ga & 0xffffffffu);               // global_addr[31:0]
  g0[3] = (unsigned)((ga >> 32) & 0x01ffffffu)        // global_addr[56:32]
        | (2u << 30);                                 // type=2 ("image")
  i32x8 g1;
  g1[0] = (int)(1u << 16);                            // data_size=1 -> 2 bytes
  g1[1] = (int)(((unsigned)Kd & 0xffffu) << 16);      // tensor_dim0[15:0]
  g1[2] = (int)((((unsigned)Kd >> 16) & 0xffffu)      // tensor_dim0[31:16]
        | (16u << 16));                               // tensor_dim1[15:0] = 16
  g1[3] = (int)(32u << 16);                           // tile_dim0 = 32
  g1[4] = 16;                                         // tile_dim1 = 16, tile_dim2 = 0
  g1[5] = Kd;                                         // tensor_dim0_stride[31:0]
  g1[6] = 0;                                          // stride hi / dim1_stride lo
  g1[7] = 0;
  i32x4 z4 = {0, 0, 0, 0};
  i32x8 z8 = {0, 0, 0, 0, 0, 0, 0, 0};
  __builtin_amdgcn_tensor_load_to_lds(g0, g1, z4, z4, z8, 0);
}
#endif

// -------------------------------------------- weight converts (one-shot)
// [E][K][N] f32 -> [E][N][K] f16
__global__ void k_cvt_t(const float* __restrict__ s, _Float16* __restrict__ d,
                        int K, int N, int total) {
  int i = blockIdx.x * 256 + threadIdx.x;
  if (i >= total) return;
  int kn = K * N;
  int e = i / kn;
  int r = i - e * kn;
  int k = r / N;
  int n = r - k * N;
  d[(size_t)e * kn + (size_t)n * K + k] = (_Float16)s[i];
}

// OIHW f32 -> [Cout][(ky*3+kx)*Cin + cin] f16
__global__ void k_cvt_conv(const float* __restrict__ s, _Float16* __restrict__ d,
                           int Cin, int Cout) {
  int i = blockIdx.x * 256 + threadIdx.x;
  if (i >= Cout * Cin * 9) return;
  int kx = i % 3; int t = i / 3;
  int ky = t % 3; t /= 3;
  int cin = t % Cin; int co = t / Cin;
  d[(size_t)co * (Cin * 9) + (ky * 3 + kx) * Cin + cin] = (_Float16)s[i];
}

// ------------------------------- conv1: 3->32, s=1, pad(1,1), out NHWC f16
__global__ void k_conv1(const float* __restrict__ img, const float* __restrict__ w,
                        const float* __restrict__ bias, _Float16* __restrict__ out) {
  int idx = blockIdx.x * 256 + threadIdx.x;   // [b][oy][ox][co] flat
  int co = idx & 31;
  int ox = (idx >> 5) & 127;
  int oy = (idx >> 12) & 127;
  int bb = idx >> 19;
  float acc = bias[co];
  const float* wp = w + co * 27;
  const float* ip = img + (size_t)bb * 3 * 16384;
  for (int ci = 0; ci < 3; ++ci)
    for (int ky = 0; ky < 3; ++ky) {
      int iy = oy + ky - 1;
      if (iy < 0 || iy > 127) continue;
      for (int kx = 0; kx < 3; ++kx) {
        int ix = ox + kx - 1;
        if (ix < 0 || ix > 127) continue;
        acc += ip[ci * 16384 + iy * 128 + ix] * wp[ci * 9 + ky * 3 + kx];
      }
    }
  out[idx] = (_Float16)fmaxf(acc, 0.f);
}

// ---- implicit-GEMM conv, 3x3, stride 2, SAME pad=(0,1), relu, NHWC f16 ----
__global__ void k_conv_wmma(const _Float16* __restrict__ X, const _Float16* __restrict__ W,
                            const float* __restrict__ bias, _Float16* __restrict__ Y,
                            int cinShift, int Cout, int Hin, int Win,
                            int hwShift, int wShift, int Kd) {
  int lane = threadIdx.x & 31;
  int wave = threadIdx.x >> 5;
  int n = ((blockIdx.y * 4 + wave) << 4) + (lane & 15);
  const _Float16* Wrow = W + (size_t)n * Kd;
  int koff = (lane >> 4) << 3;
  int Cin = 1 << cinShift;
  int hwMask = (1 << hwShift) - 1;
  int bR[4], oyR[4], oxR[4];
#pragma unroll
  for (int t = 0; t < 4; ++t) {
    int m = blockIdx.x * 64 + t * 16 + (lane & 15);
    bR[t] = m >> hwShift;
    int r = m & hwMask;
    oyR[t] = r >> wShift;
    oxR[t] = r & ((1 << wShift) - 1);
  }
  v8f acc0 = {}, acc1 = {}, acc2 = {}, acc3 = {};
  for (int kk = 0; kk < Kd; kk += 32) {
    int kb = kk + koff;
    v16h bf = ldfrag(Wrow + kb);
    v16h aT[4];
#pragma unroll
    for (int g2 = 0; g2 < 2; ++g2) {
      int kbg = kb + 16 * g2;
      int patch = kbg >> cinShift;
      int cin0 = kbg & (Cin - 1);
      int ky = (patch * 11) >> 5;
      int kx = patch - ky * 3;
#pragma unroll
      for (int t = 0; t < 4; ++t) {
        int iy = (oyR[t] << 1) + ky;
        int ix = (oxR[t] << 1) + kx;
        v8h v = {};
        if (iy < Hin && ix < Win)
          v = *(const v8h*)(X + ((((size_t)bR[t] * Hin + iy) * Win + ix) << cinShift) + cin0);
#pragma unroll
        for (int j = 0; j < 8; ++j) aT[t][g2 * 8 + j] = v[j];
      }
    }
    acc0 = wmma16(aT[0], bf, acc0);
    acc1 = wmma16(aT[1], bf, acc1);
    acc2 = wmma16(aT[2], bf, acc2);
    acc3 = wmma16(aT[3], bf, acc3);
  }
  float bv = bias[n];
  v8f accs[4] = {acc0, acc1, acc2, acc3};
#pragma unroll
  for (int t = 0; t < 4; ++t)
#pragma unroll
    for (int g = 0; g < 8; ++g) {
      int mm = blockIdx.x * 64 + t * 16 + g + ((lane >> 4) << 3);
      float v = fmaxf(accs[t][g] + bv, 0.f);
      Y[(size_t)mm * Cout + n] = (_Float16)v;
    }
}

// ---- generic WMMA GEMM: O[M,N] = A[M,K] @ Bt[N,K]^T (+bias, opt relu) ----
// B (weight) tiles are streamed LDS-side by the Tensor Data Mover, double-buffered.
__global__ void k_gemm_wmma(const _Float16* __restrict__ A, long long sA,
                            const _Float16* __restrict__ Bt, long long sB,
                            const float* __restrict__ bias, int sBias,
                            _Float16* __restrict__ O16, float* __restrict__ O32,
                            long long sO, int N, int Kd, int relu) {
#if USE_TDM
  __shared__ _Float16 btile[4][2][512];           // per-wave double buffer, 16x32 f16
#endif
  int lane = threadIdx.x & 31;
  int wave = threadIdx.x >> 5;
  int e = blockIdx.z;
  const _Float16* Ae = A + (size_t)e * sA;
  int n0 = (blockIdx.y * 4 + wave) << 4;
  int n = n0 + (lane & 15);
  const _Float16* Bbase = Bt + (size_t)e * sB + (size_t)n0 * Kd;  // 16-row tile base
  int koff = (lane >> 4) << 3;
  int ml = blockIdx.x * 64 + (lane & 15);
  const _Float16* Ar0 = Ae + (size_t)ml * Kd;
  const _Float16* Ar1 = Ar0 + (size_t)16 * Kd;
  const _Float16* Ar2 = Ar0 + (size_t)32 * Kd;
  const _Float16* Ar3 = Ar0 + (size_t)48 * Kd;
  v8f acc0 = {}, acc1 = {}, acc2 = {}, acc3 = {};
#if USE_TDM
  unsigned lds0 = (unsigned)(size_t)&btile[wave][0][0];
  unsigned lds1 = (unsigned)(size_t)&btile[wave][1][0];
  tdm_load_tile(Bbase, lds0, Kd);                 // prime chunk 0
#endif
  for (int kk = 0; kk < Kd; kk += 32) {
    int kb = kk + koff;
    v16h b;
#if USE_TDM
    unsigned cur = (kk >> 5) & 1;
    if (kk + 32 < Kd) {                           // issue next chunk, wait for this one
      tdm_load_tile(Bbase + kk + 32, cur ? lds0 : lds1, Kd);
      __builtin_amdgcn_s_wait_tensorcnt(1);
    } else {
      __builtin_amdgcn_s_wait_tensorcnt(0);
    }
    b = ldfrag(&btile[wave][cur][(lane & 15) * 32 + koff]);
#else
    __builtin_prefetch(Bbase + (size_t)(lane & 15) * Kd + kb + 128, 0, 0);
    b = ldfrag(Bbase + (size_t)(lane & 15) * Kd + kb);
#endif
    v16h a0 = ldfrag(Ar0 + kb);
    v16h a1 = ldfrag(Ar1 + kb);
    v16h a2 = ldfrag(Ar2 + kb);
    v16h a3 = ldfrag(Ar3 + kb);
    acc0 = wmma16(a0, b, acc0);
    acc1 = wmma16(a1, b, acc1);
    acc2 = wmma16(a2, b, acc2);
    acc3 = wmma16(a3, b, acc3);
  }
  float bv = bias ? bias[e * sBias + n] : 0.f;
  v8f accs[4] = {acc0, acc1, acc2, acc3};
#pragma unroll
  for (int t = 0; t < 4; ++t)
#pragma unroll
    for (int g = 0; g < 8; ++g) {
      int mm = blockIdx.x * 64 + t * 16 + g + ((lane >> 4) << 3);
      float v = accs[t][g] + bv;
      if (relu) v = fmaxf(v, 0.f);
      size_t o = (size_t)e * sO + (size_t)mm * N + n;
      if (O16) O16[o] = (_Float16)v;
      if (O32) O32[o] = v;
    }
}

// ------------------------------------------------------ global average pool
__global__ void k_gap(const _Float16* __restrict__ act4, _Float16* __restrict__ pooled) {
  int idx = blockIdx.x * 256 + threadIdx.x;
  int b = idx >> 8, c = idx & 255;
  const _Float16* p = act4 + (size_t)b * 65536 + c;
  float s = 0.f;
  for (int i = 0; i < 256; ++i) s += (float)p[i * 256];
  pooled[idx] = (_Float16)(s * (1.f / 256.f));
}

// ------------------------------------------------------ masked-mean text enc
__global__ void k_txt(const int* __restrict__ tok, const int* __restrict__ mask,
                      const float* __restrict__ embed, _Float16* __restrict__ h) {
  int b = blockIdx.x;
  int d = blockIdx.y * 256 + threadIdx.x;
  float s = 0.f, cnt = 0.f;
  for (int l = 0; l < 64; ++l) {
    float mk = (float)mask[b * 64 + l];
    cnt += mk;
    s += mk * embed[(size_t)tok[b * 64 + l] * 1024 + d];
  }
  h[(size_t)b * 1024 + d] = (_Float16)(s / fmaxf(cnt, 1.f));
}

// --------------------------------------- gate: logits, softmax, top-2, probs
__global__ void k_gate(const _Float16* __restrict__ h, const float* __restrict__ gw,
                       const float* __restrict__ gb, int* __restrict__ topidx,
                       float* __restrict__ topw, float* __restrict__ probs) {
  int b = threadIdx.x;
  float lg[8];
#pragma unroll
  for (int e = 0; e < 8; ++e) lg[e] = gb[e];
  for (int d = 0; d < 1024; ++d) {
    float x = (float)h[(size_t)b * 1024 + d];
#pragma unroll
    for (int e = 0; e < 8; ++e) lg[e] += x * gw[d * 8 + e];
  }
  float mx = lg[0];
#pragma unroll
  for (int e = 1; e < 8; ++e) mx = fmaxf(mx, lg[e]);
  float p[8], sm = 0.f;
#pragma unroll
  for (int e = 0; e < 8; ++e) { p[e] = __expf(lg[e] - mx); sm += p[e]; }
#pragma unroll
  for (int e = 0; e < 8; ++e) { p[e] /= sm; probs[b * 8 + e] = p[e]; }
  int i0 = 0;
#pragma unroll
  for (int e = 1; e < 8; ++e) if (p[e] > p[i0]) i0 = e;
  int i1 = (i0 == 0) ? 1 : 0;
#pragma unroll
  for (int e = 0; e < 8; ++e) if (e != i0 && p[e] > p[i1]) i1 = e;
  float s2 = p[i0] + p[i1] + 1e-9f;
  topidx[b * 2] = i0; topidx[b * 2 + 1] = i1;
  topw[b * 2] = p[i0] / s2; topw[b * 2 + 1] = p[i1] / s2;
}

// --------------------------------------------- top-2 weighted expert combine
__global__ void k_combine(const _Float16* __restrict__ Yexp, const int* __restrict__ topidx,
                          const float* __restrict__ topw, _Float16* __restrict__ z) {
  int idx = blockIdx.x * 256 + threadIdx.x;
  int b = idx >> 10, d = idx & 1023;
  int i0 = topidx[b * 2], i1 = topidx[b * 2 + 1];
  float v = topw[b * 2]     * (float)Yexp[((size_t)i0 * 256 + b) * 1024 + d]
          + topw[b * 2 + 1] * (float)Yexp[((size_t)i1 * 256 + b) * 1024 + d];
  z[idx] = (_Float16)v;
}

// ------------------------------------------------------------- L2 normalize
__global__ void k_norm(const float* __restrict__ zp, _Float16* __restrict__ e) {
  __shared__ float red[256];
  int b = blockIdx.x, t = threadIdx.x;
  float s = 0.f;
  for (int i = 0; i < 4; ++i) { float v = zp[b * 1024 + t + i * 256]; s += v * v; }
  red[t] = s; __syncthreads();
  for (int off = 128; off; off >>= 1) {
    if (t < off) red[t] += red[t + off];
    __syncthreads();
  }
  float inv = 1.f / fmaxf(sqrtf(red[0]), 1e-12f);
  for (int i = 0; i < 4; ++i)
    e[b * 1024 + t + i * 256] = (_Float16)(zp[b * 1024 + t + i * 256] * inv);
}

// ------------------------------------- logits = scale * Ei @ Et^T (and ^T out)
__global__ void k_logits(const _Float16* __restrict__ Ei, const _Float16* __restrict__ Et,
                         const float* __restrict__ lsc, float* __restrict__ out) {
  int lane = threadIdx.x & 31, wave = threadIdx.x >> 5;
  int n = ((blockIdx.y * 4 + wave) << 4) + (lane & 15);
  const _Float16* Brow = Et + (size_t)n * 1024;
  int koff = (lane >> 4) << 3;
  int ml = blockIdx.x * 64 + (lane & 15);
  const _Float16* Ar0 = Ei + (size_t)ml * 1024;
  v8f acc0 = {}, acc1 = {}, acc2 = {}, acc3 = {};
  for (int kk = 0; kk < 1024; kk += 32) {
    int kb = kk + koff;
    v16h b  = ldfrag(Brow + kb);
    v16h a0 = ldfrag(Ar0 + kb);
    v16h a1 = ldfrag(Ar0 + 16 * 1024 + kb);
    v16h a2 = ldfrag(Ar0 + 32 * 1024 + kb);
    v16h a3 = ldfrag(Ar0 + 48 * 1024 + kb);
    acc0 = wmma16(a0, b, acc0);
    acc1 = wmma16(a1, b, acc1);
    acc2 = wmma16(a2, b, acc2);
    acc3 = wmma16(a3, b, acc3);
  }
  float scale = fminf(__expf(lsc[0]), 100.f);
  v8f accs[4] = {acc0, acc1, acc2, acc3};
#pragma unroll
  for (int t = 0; t < 4; ++t)
#pragma unroll
    for (int g = 0; g < 8; ++g) {
      int mm = blockIdx.x * 64 + t * 16 + g + ((lane >> 4) << 3);
      float v = scale * accs[t][g];
      out[mm * 256 + n] = v;
      out[65536 + n * 256 + mm] = v;
    }
}

// ------------------------------------------------- deterministic KL aux loss
__global__ void k_aux(const float* __restrict__ probs, float* __restrict__ outAux) {
  if (threadIdx.x == 0 && blockIdx.x == 0) {
    float total = 0.f;
    for (int mdl = 0; mdl < 2; ++mdl) {
      float aux = 0.f;
      for (int e = 0; e < 8; ++e) {
        float mp = 0.f;
        for (int b = 0; b < 256; ++b) mp += probs[mdl * 2048 + b * 8 + e];
        mp *= (1.f / 256.f);
        aux += 0.125f * (logf(0.125f) - logf(mp));
      }
      total += aux * 0.125f;
    }
    *outAux = total * 0.5f;
  }
}

// ===========================================================================
extern "C" void kernel_launch(void* const* d_in, const int* in_sizes, int n_in,
                              void* d_out, int out_size, void* d_ws, size_t ws_size,
                              hipStream_t stream) {
  (void)in_sizes; (void)n_in; (void)out_size; (void)ws_size;
  const float* img   = (const float*)d_in[0];
  const int*   tok   = (const int*)  d_in[1];
  const int*   mask  = (const int*)  d_in[2];
  const float* cw1   = (const float*)d_in[3];
  const float* cb1   = (const float*)d_in[4];
  const float* cw2   = (const float*)d_in[5];
  const float* cb2   = (const float*)d_in[6];
  const float* cw3   = (const float*)d_in[7];
  const float* cb3   = (const float*)d_in[8];
  const float* cw4   = (const float*)d_in[9];
  const float* cb4   = (const float*)d_in[10];
  const float* epw   = (const float*)d_in[11];
  const float* epb   = (const float*)d_in[12];
  const float* emb   = (const float*)d_in[13];
  const float* gw    = (const float*)d_in[14];
  const float* gb    = (const float*)d_in[15];
  const float* ew1   = (const float*)d_in[16];
  const float* eb1   = (const float*)d_in[17];
  const float* ew2   = (const float*)d_in[18];
  const float* eb2   = (const float*)d_in[19];
  const float* ipw   = (const float*)d_in[20];
  const float* tpw   = (const float*)d_in[21];
  const float* lsc   = (const float*)d_in[22];
  float* out = (float*)d_out;

  char* w = (char*)d_ws;
  auto alloc = [&](size_t bytes) { void* p = (void*)w; w += (bytes + 255) & ~(size_t)255; return p; };
  _Float16* w2h  = (_Float16*)alloc(18432  * 2);
  _Float16* w3h  = (_Float16*)alloc(73728  * 2);
  _Float16* w4h  = (_Float16*)alloc(294912 * 2);
  _Float16* wph  = (_Float16*)alloc(262144 * 2);
  _Float16* w1h  = (_Float16*)alloc((size_t)4194304 * 2);
  _Float16* w2eh = (_Float16*)alloc((size_t)4194304 * 2);
  _Float16* wpi  = (_Float16*)alloc((size_t)1048576 * 2);
  _Float16* wpt  = (_Float16*)alloc((size_t)1048576 * 2);
  _Float16* act1 = (_Float16*)alloc((size_t)134217728 * 2); // B*128*128*32 NHWC
  _Float16* act2 = (_Float16*)alloc((size_t)67108864  * 2); // B*64*64*64
  _Float16* act3 = act1;
  _Float16* act4 = act2;
  _Float16* pooled = (_Float16*)alloc(65536  * 2);
  _Float16* h_img  = (_Float16*)alloc(262144 * 2);
  _Float16* h_txt  = (_Float16*)alloc(262144 * 2);
  _Float16* hidden = (_Float16*)alloc((size_t)1048576 * 2);
  _Float16* Yexp   = (_Float16*)alloc((size_t)2097152 * 2);
  _Float16* zbuf   = (_Float16*)alloc(262144 * 2);
  _Float16* e_img  = (_Float16*)alloc(262144 * 2);
  _Float16* e_txt  = (_Float16*)alloc(262144 * 2);
  float* zproj  = (float*)alloc(262144 * 4);
  float* probs  = (float*)alloc(4096 * 4);
  int*   topidx = (int*)  alloc(512 * 4);
  float* topw   = (float*)alloc(512 * 4);

  auto cvtT = [&](const float* s, _Float16* d, int K, int N, int total) {
    k_cvt_t<<<dim3((total + 255) / 256), dim3(256), 0, stream>>>(s, d, K, N, total);
  };
  k_cvt_conv<<<dim3(72, 1), dim3(256), 0, stream>>>(cw2, w2h, 32, 64);
  k_cvt_conv<<<dim3(288, 1), dim3(256), 0, stream>>>(cw3, w3h, 64, 128);
  k_cvt_conv<<<dim3(1152, 1), dim3(256), 0, stream>>>(cw4, w4h, 128, 256);
  cvtT(epw, wph, 256, 1024, 262144);
  cvtT(ew1, w1h, 1024, 512, 4194304);
  cvtT(ew2, w2eh, 512, 1024, 4194304);
  cvtT(ipw, wpi, 1024, 1024, 1048576);
  cvtT(tpw, wpt, 1024, 1024, 1048576);

  // ---- image tower ------------------------------------------------------
  k_conv1<<<dim3(524288), dim3(256), 0, stream>>>(img, cw1, cb1, act1);
  k_conv_wmma<<<dim3(16384, 1), dim3(128), 0, stream>>>(act1, w2h, cb2, act2,
      5, 64, 128, 128, 12, 6, 288);
  k_conv_wmma<<<dim3(4096, 2), dim3(128), 0, stream>>>(act2, w3h, cb3, act3,
      6, 128, 64, 64, 10, 5, 576);
  k_conv_wmma<<<dim3(1024, 4), dim3(128), 0, stream>>>(act3, w4h, cb4, act4,
      7, 256, 32, 32, 8, 4, 1152);
  k_gap<<<dim3(256), dim3(256), 0, stream>>>(act4, pooled);
  k_gemm_wmma<<<dim3(4, 16, 1), dim3(128), 0, stream>>>(pooled, 0, wph, 0,
      epb, 0, h_img, nullptr, 0, 1024, 256, 0);

  // ---- text tower -------------------------------------------------------
  k_txt<<<dim3(256, 4), dim3(256), 0, stream>>>(tok, mask, emb, h_txt);

  // ---- MoE + projection + normalize per modality ------------------------
  for (int mdl = 0; mdl < 2; ++mdl) {
    _Float16* h    = mdl ? h_txt : h_img;
    _Float16* proj = mdl ? wpt : wpi;
    _Float16* eout = mdl ? e_txt : e_img;
    k_gate<<<dim3(1), dim3(256), 0, stream>>>(h, gw, gb, topidx, topw,
        probs + mdl * 2048);
    k_gemm_wmma<<<dim3(4, 8, 8), dim3(128), 0, stream>>>(h, 0, w1h, 524288,
        eb1, 512, hidden, nullptr, 131072, 512, 1024, 1);
    k_gemm_wmma<<<dim3(4, 16, 8), dim3(128), 0, stream>>>(hidden, 131072,
        w2eh, 524288, eb2, 1024, Yexp, nullptr, 262144, 1024, 512, 0);
    k_combine<<<dim3(1024), dim3(256), 0, stream>>>(Yexp, topidx, topw, zbuf);
    k_gemm_wmma<<<dim3(4, 16, 1), dim3(128), 0, stream>>>(zbuf, 0, proj, 0,
        nullptr, 0, nullptr, zproj, 0, 1024, 1024, 0);
    k_norm<<<dim3(256), dim3(256), 0, stream>>>(zproj, eout);
  }

  // ---- contrastive logits + aux loss ------------------------------------
  k_logits<<<dim3(4, 4), dim3(128), 0, stream>>>(e_img, e_txt, lsc, out);
  k_aux<<<dim3(1), dim3(32), 0, stream>>>(probs, out + 131072);
}
